// PointEncoderRaw_74680891342897
// MI455X (gfx1250) — matrix operand order repaired and checked by reading, not compiled, not run on hardware
//
#include <hip/hip_runtime.h>
#include <hip/hip_bf16.h>

#define B_ 8
#define N_ 2048
#define K_ 16
#define OUT_ 128
#define RANK_ 32
#define BN_ (B_ * N_)

typedef __attribute__((ext_vector_type(8)))  float  v8f;
typedef __attribute__((ext_vector_type(16))) __bf16 v16bf;

union BF16x16 {
    v16bf v;
    __hip_bfloat16 h[16];
    int4 i[2];
};

__device__ __forceinline__ unsigned enc_f(float f) {
    unsigned u = __float_as_uint(f);
    return (u & 0x80000000u) ? ~u : (u | 0x80000000u);
}
__device__ __forceinline__ float dec_f(unsigned k) {
    unsigned u = (k & 0x80000000u) ? (k & 0x7FFFFFFFu) : ~k;
    return __uint_as_float(u);
}

// ---------------------------------------------------------------- top-k(16)
// one block (64 threads) per (b,n) row; smallest-dist-first, stable by index.
__global__ __launch_bounds__(64) void topk_kernel(
    const float* __restrict__ dist, int* __restrict__ nbrs)
{
    __shared__ float sd[N_];
    __shared__ unsigned char used[N_];
    __shared__ unsigned long long red[64];
    const int row = blockIdx.x;
    const float* d = dist + (size_t)row * N_;
    for (int i = threadIdx.x; i < N_; i += 64) { sd[i] = d[i]; used[i] = 0; }
    __syncthreads();
    for (int sel = 0; sel < K_; ++sel) {
        unsigned long long best = ~0ull;
        for (int i = threadIdx.x; i < N_; i += 64) {
            if (!used[i]) {
                unsigned long long cand =
                    ((unsigned long long)enc_f(sd[i]) << 32) | (unsigned)i;
                best = (cand < best) ? cand : best;
            }
        }
        red[threadIdx.x] = best;
        __syncthreads();
        for (int off = 32; off > 0; off >>= 1) {
            if (threadIdx.x < off) {
                unsigned long long o = red[threadIdx.x + off];
                if (o < red[threadIdx.x]) red[threadIdx.x] = o;
            }
            __syncthreads();
        }
        int widx = (int)(red[0] & 0xffffffffu);
        if (threadIdx.x == 0) nbrs[(size_t)row * K_ + sel] = widx;
        used[widx] = 1;
        __syncthreads();
    }
}

// ---------------------------------------------------------------- geometry
// one wave per point, lanes 0..15 handle the K=16 neighbors.
__global__ __launch_bounds__(128) void geom_kernel(
    const float* __restrict__ pc, const float* __restrict__ nrm,
    const int* __restrict__ nbrs,
    float* __restrict__ rifeat, float* __restrict__ feat0)
{
    const int p = blockIdx.x * 4 + (threadIdx.x >> 5);
    const int lane = threadIdx.x & 31;
    const int b = p / N_, n = p % N_;
    const float* pcb = pc + (size_t)b * N_ * 3;
    const float* nb  = nrm + (size_t)b * N_ * 3;
    const int k = lane;
    int idx = 0;
    float nx = 0.f, ny = 0.f, nz = 0.f;
    if (k < K_) {
        idx = nbrs[(size_t)p * K_ + k];
        nx = pcb[idx * 3 + 0]; ny = pcb[idx * 3 + 1]; nz = pcb[idx * 3 + 2];
    }
    float sx = nx, sy = ny, sz = nz;
    for (int m = 8; m >= 1; m >>= 1) {
        sx += __shfl_xor(sx, m, 32);
        sy += __shfl_xor(sy, m, 32);
        sz += __shfl_xor(sz, m, 32);
    }
    const float mx = sx * (1.f / 16.f), my = sy * (1.f / 16.f), mz = sz * (1.f / 16.f);
    const float cx = pcb[n * 3 + 0], cy = pcb[n * 3 + 1], cz = pcb[n * 3 + 2];
    if (k < K_) {
        const float l1x = mx - nx, l1y = my - ny, l1z = mz - nz;
        const float l2x = nx - cx, l2y = ny - cy, l2z = nz - cz;
        const float l3x = cx - mx, l3y = cy - my, l3z = cz - mz;
        const float l1n = sqrtf(l1x * l1x + l1y * l1y + l1z * l1z);
        const float l2n = sqrtf(l2x * l2x + l2y * l2y + l2z * l2z);
        const float l3n = sqrtf(l3x * l3x + l3y * l3y + l3z * l3z);
        const float t1 = (l1x * l2x + l1y * l2y + l1z * l2z) / (l1n * l2n + 1e-7f);
        const float t2 = (l2x * l3x + l2y * l3y + l2z * l3z) / (l2n * l3n + 1e-7f);
        const float t3 = (l3x * l1x + l3y * l1y + l3z * l1z) / (l3n * l1n + 1e-7f);
        float* rf = rifeat + ((size_t)p * K_ + k) * 6;
        rf[0] = l1n; rf[1] = l2n; rf[2] = l3n; rf[3] = t1; rf[4] = t2; rf[5] = t3;
        const float cosv = nb[idx * 3 + 0] * nb[n * 3 + 0] +
                           nb[idx * 3 + 1] * nb[n * 3 + 1] +
                           nb[idx * 3 + 2] * nb[n * 3 + 2];
        float* f0 = feat0 + ((size_t)p * K_ + k) * 2;
        f0[0] = l2n; f0[1] = cosv;
    }
}

// ---------------------------------------------------------------- kernel MLP
// per-row 6->32 (LN,relu) ->64 (LN,relu) ->32 ; weights staged in LDS.
__global__ __launch_bounds__(256) void mlp_kernel(
    const float* __restrict__ rifeat,
    const float* __restrict__ w1, const float* __restrict__ b1,
    const float* __restrict__ g1, const float* __restrict__ gb1,
    const float* __restrict__ w2, const float* __restrict__ b2,
    const float* __restrict__ g2, const float* __restrict__ gb2,
    const float* __restrict__ w3, const float* __restrict__ b3,
    float* __restrict__ kern, int rows)
{
    __shared__ float sw1[6 * 32], sb1[32], sg1[32], sgb1[32];
    __shared__ float sw2[32 * 64], sb2[64], sg2[64], sgb2[64];
    __shared__ float sw3[64 * 32], sb3[32];
    for (int i = threadIdx.x; i < 6 * 32; i += 256) sw1[i] = w1[i];
    for (int i = threadIdx.x; i < 32; i += 256) {
        sb1[i] = b1[i]; sg1[i] = g1[i]; sgb1[i] = gb1[i]; sb3[i] = b3[i];
    }
    for (int i = threadIdx.x; i < 32 * 64; i += 256) sw2[i] = w2[i];
    for (int i = threadIdx.x; i < 64; i += 256) {
        sb2[i] = b2[i]; sg2[i] = g2[i]; sgb2[i] = gb2[i];
    }
    for (int i = threadIdx.x; i < 64 * 32; i += 256) sw3[i] = w3[i];
    __syncthreads();

    const int r = blockIdx.x * 256 + threadIdx.x;
    if (r >= rows) return;
    float x[6];
#pragma unroll
    for (int c = 0; c < 6; ++c) x[c] = rifeat[(size_t)r * 6 + c];

    float h1[32];
    for (int j = 0; j < 32; ++j) {
        float s = sb1[j];
#pragma unroll
        for (int c = 0; c < 6; ++c) s += x[c] * sw1[c * 32 + j];
        h1[j] = s;
    }
    float m = 0.f;
    for (int j = 0; j < 32; ++j) m += h1[j];
    m *= (1.f / 32.f);
    float v = 0.f;
    for (int j = 0; j < 32; ++j) { float d = h1[j] - m; v += d * d; }
    v *= (1.f / 32.f);
    float inv = rsqrtf(v + 1e-5f);
    for (int j = 0; j < 32; ++j) {
        float y = (h1[j] - m) * inv * sg1[j] + sgb1[j];
        h1[j] = y > 0.f ? y : 0.f;
    }

    float h2[64];
    for (int j = 0; j < 64; ++j) {
        float s = sb2[j];
#pragma unroll
        for (int c = 0; c < 32; ++c) s += h1[c] * sw2[c * 64 + j];
        h2[j] = s;
    }
    m = 0.f;
    for (int j = 0; j < 64; ++j) m += h2[j];
    m *= (1.f / 64.f);
    v = 0.f;
    for (int j = 0; j < 64; ++j) { float d = h2[j] - m; v += d * d; }
    v *= (1.f / 64.f);
    inv = rsqrtf(v + 1e-5f);
    for (int j = 0; j < 64; ++j) {
        float y = (h2[j] - m) * inv * sg2[j] + sgb2[j];
        h2[j] = y > 0.f ? y : 0.f;
    }

    for (int j = 0; j < 32; ++j) {
        float s = sb3[j];
#pragma unroll
        for (int c = 0; c < 64; ++c) s += h2[c] * sw3[c * 32 + j];
        kern[(size_t)r * 32 + j] = s;
    }
}

// ------------------------------------------------- einsum contraction, layer0
// contracted[p, r*2+i] = sum_k kern[p,k,r] * feat0[p,k,i]  (bf16 out)
__global__ __launch_bounds__(256) void contract0_kernel(
    const float* __restrict__ kern, const float* __restrict__ feat0,
    __hip_bfloat16* __restrict__ Y)
{
    const int p = blockIdx.x * 8 + (threadIdx.x >> 5);
    const int r = threadIdx.x & 31;
    const float* kp = kern + (size_t)p * K_ * 32;
    const float* fp = feat0 + (size_t)p * K_ * 2;
    float c0 = 0.f, c1 = 0.f;
#pragma unroll
    for (int k = 0; k < K_; ++k) {
        float kv = kp[k * 32 + r];
        c0 += kv * fp[k * 2 + 0];
        c1 += kv * fp[k * 2 + 1];
    }
    Y[(size_t)p * 64 + r * 2 + 0] = __float2bfloat16(c0);
    Y[(size_t)p * 64 + r * 2 + 1] = __float2bfloat16(c1);
}

// ------------------------------------------------- einsum contraction, layer1
// contracted[p, r*160+i] = sum_k kern[p,k,r] * feat160[b, nbr(p,k), i]
__global__ __launch_bounds__(256) void contract1_kernel(
    const float* __restrict__ kern, const float* __restrict__ feat160,
    const int* __restrict__ nbrs, __hip_bfloat16* __restrict__ Y)
{
    __shared__ float sf[K_ * 160];
    __shared__ float sk[K_ * 32];
    const int p = blockIdx.x;
    const int b = p / N_;
    for (int i = threadIdx.x; i < K_ * 160; i += 256) {
        int k = i / 160, c = i % 160;
        int idx = nbrs[(size_t)p * K_ + k];
        sf[i] = feat160[((size_t)b * N_ + idx) * 160 + c];
    }
    for (int i = threadIdx.x; i < K_ * 32; i += 256)
        sk[i] = kern[(size_t)p * K_ * 32 + i];
    __syncthreads();
    for (int o = threadIdx.x; o < 5120; o += 256) {
        int r = o / 160, c = o % 160;
        float s = 0.f;
#pragma unroll
        for (int k = 0; k < K_; ++k) s += sk[k * 32 + r] * sf[k * 160 + c];
        Y[(size_t)p * 5120 + o] = __float2bfloat16(s);
    }
}

// ----------------------------------------- fp32 -> bf16, B-fragment swizzle
// dst layout: [kblk][tile(8)][lane(32)][e(16)] so a lane's whole WMMA B
// fragment is 32 contiguous bytes.  B layout: lane<16 -> col=lane, K=e;
// lane>=16 -> col=lane-16, K=16+e (within the 32-deep k-slab).
__global__ __launch_bounds__(256) void f2bf_swz_kernel(
    const float* __restrict__ W, __hip_bfloat16* __restrict__ dst, int Kdim)
{
    int i = blockIdx.x * 256 + threadIdx.x;
    if (i >= Kdim * 128) return;
    int e    = i & 15;
    int lane = (i >> 4) & 31;
    int tile = (i >> 9) & 7;
    int kblk = i >> 12;
    int n = tile * 16 + (lane & 15);
    int k = kblk * 32 + (lane >> 4) * 16 + e;
    dst[i] = __float2bfloat16(W[(size_t)k * 128 + n]);
}

// ---------------------------------------------------------------- WMMA GEMM
// out[M x 128] = Y[M x Kdim](bf16) @ W[Kdim x 128](bf16, pre-swizzled) + bias.
// block = 256 (8 waves) -> 128 rows x 128 cols; wave: 32 rows x 64 cols
// (2 A-tiles x 4 B-tiles = 8 WMMAs / k-step).  No LDS, no barriers: B
// fragments stream as coalesced b128 loads from the L2-resident swizzled W.
__global__ __launch_bounds__(256) void gemm_bf16_kernel(
    const __hip_bfloat16* __restrict__ Y, int Kdim,
    const __hip_bfloat16* __restrict__ Wswz,
    const float* __restrict__ bias, float* __restrict__ out)
{
    const int tid = threadIdx.x;
    const int wave = tid >> 5, lane = tid & 31;
    const int rowBase = blockIdx.x * 128 + (wave >> 1) * 32;
    const int colHalf = wave & 1;
    const int laneHi = lane >> 4;
    const int lm = lane & 15;

    v8f acc[2][4];
#pragma unroll
    for (int s = 0; s < 2; ++s)
#pragma unroll
        for (int t = 0; t < 4; ++t)
#pragma unroll
            for (int e = 0; e < 8; ++e) acc[s][t][e] = 0.f;

    const size_t rowOff0 = (size_t)(rowBase + lm) * Kdim;
    const size_t rowOff1 = (size_t)(rowBase + 16 + lm) * Kdim;

    for (int k0 = 0; k0 < Kdim; k0 += 32) {
        // A fragments (16x32): contiguous k order, same as the swizzled B.
        // lane<16: row=lm, K = {k0..k0+15}
        // lane>=16: row=lm, K = {k0+16..k0+31}
        const int kb = k0 + laneHi * 16;
        BF16x16 a0, a1;
        a0.i[0] = *reinterpret_cast<const int4*>(Y + rowOff0 + kb);
        a0.i[1] = *reinterpret_cast<const int4*>(Y + rowOff0 + kb + 8);
        a1.i[0] = *reinterpret_cast<const int4*>(Y + rowOff1 + kb);
        a1.i[1] = *reinterpret_cast<const int4*>(Y + rowOff1 + kb + 8);
        if (k0 + 32 < Kdim) {          // pull next k-slab (global_prefetch_b8)
            __builtin_prefetch(Y + rowOff0 + kb + 32, 0, 3);
            __builtin_prefetch(Y + rowOff1 + kb + 32, 0, 3);
        }

        const int kblk = k0 >> 5;
#pragma unroll
        for (int t = 0; t < 4; ++t) {
            BF16x16 bf;
            const int4* bp = reinterpret_cast<const int4*>(
                Wswz + ((((size_t)kblk * 8 + colHalf * 4 + t) * 32 + lane) << 4));
            bf.i[0] = bp[0];
            bf.i[1] = bp[1];
            acc[0][t] = __builtin_amdgcn_wmma_f32_16x16x32_bf16(
                false, a0.v, false, bf.v, (short)0, acc[0][t], false, false);
            acc[1][t] = __builtin_amdgcn_wmma_f32_16x16x32_bf16(
                false, a1.v, false, bf.v, (short)0, acc[1][t], false, false);
        }
    }

    // C layout: VGPR v -> row = strip + v + (laneHi?8:0), col = colHalf*64+t*16+lm
#pragma unroll
    for (int s = 0; s < 2; ++s) {
        const int row0 = rowBase + s * 16 + laneHi * 8;
#pragma unroll
        for (int t = 0; t < 4; ++t) {
            const int col = colHalf * 64 + t * 16 + lm;
            const float bv = bias[col];
#pragma unroll
            for (int v = 0; v < 8; ++v)
                out[(size_t)(row0 + v) * 128 + col] = acc[s][t][v] + bv;
        }
    }
}

// --------------------------------------------------------- LN + aggr-linear
// one wave per row: LayerNorm(128) -> dst[0..127]; tran(32) -> atomicMax glob.
__global__ __launch_bounds__(32) void ln_aggr_kernel(
    const float* __restrict__ pre, const float* __restrict__ g,
    const float* __restrict__ bb, const float* __restrict__ aggw,
    const float* __restrict__ aggb, float* __restrict__ dst, int ldDst,
    unsigned* __restrict__ glob)
{
    __shared__ float y[128];
    const int row = blockIdx.x;
    const int lane = threadIdx.x;
    float x[4];
#pragma unroll
    for (int c = 0; c < 4; ++c) x[c] = pre[(size_t)row * 128 + lane * 4 + c];
    float s = x[0] + x[1] + x[2] + x[3];
    float sq = x[0] * x[0] + x[1] * x[1] + x[2] * x[2] + x[3] * x[3];
    for (int m = 1; m < 32; m <<= 1) {
        s += __shfl_xor(s, m, 32);
        sq += __shfl_xor(sq, m, 32);
    }
    const float mean = s * (1.f / 128.f);
    const float var = sq * (1.f / 128.f) - mean * mean;
    const float inv = rsqrtf(var + 1e-5f);
#pragma unroll
    for (int c = 0; c < 4; ++c) {
        float yy = (x[c] - mean) * inv * g[lane * 4 + c] + bb[lane * 4 + c];
        y[lane * 4 + c] = yy;
        dst[(size_t)row * ldDst + lane * 4 + c] = yy;
    }
    __syncthreads();
    const int b = row / N_;
    float t = aggb[lane];
    for (int c = 0; c < 128; ++c) t += y[c] * aggw[c * 32 + lane];
    atomicMax(glob + b * 32 + lane, enc_f(t));
}

__global__ __launch_bounds__(256) void bcast_kernel(
    float* __restrict__ dst, int ldDst, const unsigned* __restrict__ glob)
{
    int i = blockIdx.x * 256 + threadIdx.x;
    if (i >= BN_ * 32) return;
    int row = i >> 5, j = i & 31;
    int b = row / N_;
    dst[(size_t)row * ldDst + 128 + j] = dec_f(glob[b * 32 + j]);
}

__global__ __launch_bounds__(256) void init_glob_kernel(unsigned* g, int n)
{
    int i = blockIdx.x * 256 + threadIdx.x;
    if (i < n) g[i] = 0u;
}

// ---------------------------------------------------------------- launcher
extern "C" void kernel_launch(void* const* d_in, const int* in_sizes, int n_in,
                              void* d_out, int out_size, void* d_ws, size_t ws_size,
                              hipStream_t stream)
{
    const float* pc   = (const float*)d_in[0];
    const float* pcn  = (const float*)d_in[1];
    const float* dist = (const float*)d_in[2];
    // params flat order per layer (16 each): w1,b1,g1,gb1,w2,b2,g2,gb2,w3,b3,
    //                                        out_w,out_b,ln_g,ln_b,agg_w,agg_b
    const float* P[32];
    for (int i = 0; i < 32; ++i) P[i] = (const float*)d_in[3 + i];
    const float** L0 = P;
    const float** L1 = P + 16;

    // workspace carve-out
    char* w = (char*)d_ws;
    auto carve = [&](size_t bytes) -> char* {
        char* p = w;
        w += (bytes + 255) & ~(size_t)255;
        return p;
    };
    int*   nbrs    = (int*)  carve((size_t)BN_ * K_ * 4);
    float* rifeat  = (float*)carve((size_t)BN_ * K_ * 6 * 4);
    float* feat0   = (float*)carve((size_t)BN_ * K_ * 2 * 4);
    float* kern    = (float*)carve((size_t)BN_ * K_ * 32 * 4);
    __hip_bfloat16* Y0  = (__hip_bfloat16*)carve((size_t)BN_ * 64 * 2);
    __hip_bfloat16* Y1  = (__hip_bfloat16*)carve((size_t)BN_ * 5120 * 2);
    __hip_bfloat16* w0b = (__hip_bfloat16*)carve((size_t)64 * 128 * 2);
    __hip_bfloat16* w1b = (__hip_bfloat16*)carve((size_t)5120 * 128 * 2);
    float* preLN   = (float*)carve((size_t)BN_ * 128 * 4);
    float* feat160 = (float*)carve((size_t)BN_ * 160 * 4);
    unsigned* glob0 = (unsigned*)carve((size_t)B_ * 32 * 4);
    unsigned* glob1 = (unsigned*)carve((size_t)B_ * 32 * 4);

    float* outf = (float*)d_out;

    // 0. init global-max accumulators (ws is not re-poisoned between replays)
    init_glob_kernel<<<1, 256, 0, stream>>>(glob0, B_ * 32);
    init_glob_kernel<<<1, 256, 0, stream>>>(glob1, B_ * 32);

    // 1. top-k neighbors (bandwidth floor: one pass over 128 MB dist)
    topk_kernel<<<BN_, 64, 0, stream>>>(dist, nbrs);

    // 2. rotation-invariant features + feat0 (shared by both layers)
    geom_kernel<<<BN_ / 4, 128, 0, stream>>>(pc, pcn, nbrs, rifeat, feat0);

    // 3. weights -> bf16, pre-swizzled into WMMA B-fragment layout
    f2bf_swz_kernel<<<(64 * 128 + 255) / 256, 256, 0, stream>>>(L0[10], w0b, 64);
    f2bf_swz_kernel<<<(5120 * 128 + 255) / 256, 256, 0, stream>>>(L1[10], w1b, 5120);

    // ---- layer 0 ----
    mlp_kernel<<<(BN_ * K_) / 256, 256, 0, stream>>>(
        rifeat, L0[0], L0[1], L0[2], L0[3], L0[4], L0[5], L0[6], L0[7],
        L0[8], L0[9], kern, BN_ * K_);
    contract0_kernel<<<BN_ / 8, 256, 0, stream>>>(kern, feat0, Y0);
    gemm_bf16_kernel<<<BN_ / 128, 256, 0, stream>>>(Y0, 64, w0b, L0[11], preLN);
    ln_aggr_kernel<<<BN_, 32, 0, stream>>>(
        preLN, L0[12], L0[13], L0[14], L0[15], feat160, 160, glob0);
    bcast_kernel<<<(BN_ * 32) / 256, 256, 0, stream>>>(feat160, 160, glob0);

    // ---- layer 1 ----
    mlp_kernel<<<(BN_ * K_) / 256, 256, 0, stream>>>(
        rifeat, L1[0], L1[1], L1[2], L1[3], L1[4], L1[5], L1[6], L1[7],
        L1[8], L1[9], kern, BN_ * K_);
    contract1_kernel<<<BN_, 256, 0, stream>>>(kern, feat160, nbrs, Y1);
    gemm_bf16_kernel<<<BN_ / 128, 256, 0, stream>>>(Y1, 5120, w1b, L1[11], preLN);
    ln_aggr_kernel<<<BN_, 32, 0, stream>>>(
        preLN, L1[12], L1[13], L1[14], L1[15], outf, 160, glob1);
    bcast_kernel<<<(BN_ * 32) / 256, 256, 0, stream>>>(outf, 160, glob1);

    (void)in_sizes; (void)n_in; (void)out_size; (void)ws_size;
}
